// EncoderLayer_52613349376211
// MI455X (gfx1250) — compile-verified, matrix operand order checked by reading, not run on hardware
//
#include <hip/hip_runtime.h>
#include <hip/hip_bf16.h>
#include <cstdint>
#include <cstddef>

// ---------------------------------------------------------------- types
typedef _Float16 half_t;
typedef __attribute__((ext_vector_type(16))) _Float16 v16h;
typedef __attribute__((ext_vector_type(8)))  _Float16 v8h;
typedef __attribute__((ext_vector_type(4)))  _Float16 v4h;
typedef __attribute__((ext_vector_type(8)))  float    v8f;
typedef __attribute__((ext_vector_type(4)))  float    v4f;

#define BATCH  4
#define SEQ    4096
#define DMODEL 1024
#define NHEAD  16
#define DHEAD  64
#define DFF    4096
#define MFEAT  64

// gfx1250 async global->LDS copy path (ASYNCcnt-tracked, bypasses VGPRs).
#define USE_ASYNC_LDS 1

static __device__ __forceinline__ void copy16_g2l(const void* gptr, void* lptr) {
#if USE_ASYNC_LDS
  // VDST = LDS byte address (low 32 bits of the generic LDS pointer),
  // VADDR = 64-bit global address, GV addressing mode.
  uint32_t lds = (uint32_t)(uintptr_t)lptr;
  asm volatile("global_load_async_to_lds_b128 %0, %1, off"
               :: "v"(lds), "v"(gptr) : "memory");
#else
  *(v8h*)lptr = *(const v8h*)gptr;
#endif
}

static __device__ __forceinline__ void async_wait_all() {
#if USE_ASYNC_LDS
#if __has_builtin(__builtin_amdgcn_s_wait_asynccnt)
  __builtin_amdgcn_s_wait_asynccnt(0);
#else
  asm volatile("s_wait_asynccnt 0" ::: "memory");
#endif
#endif
}

static __device__ __forceinline__ v16h mkfrag(v8h lo, v8h hi) {
  union { v16h v; v8h h[2]; } u;
  u.h[0] = lo; u.h[1] = hi;
  return u.v;
}

static __device__ __forceinline__ v8f wmma32(v16h a, v16h b, v8f c) {
  // D(16x16,f32) = A(16x32,f16) x B(32x16,f16) + C
  return __builtin_amdgcn_wmma_f32_16x16x32_f16(false, a, false, b, (short)0, c,
                                                false, false);
}

static __device__ __forceinline__ void atomicMaxF(float* addr, float val) {
  int* ai = (int*)addr;
  int old = __float_as_int(*addr);
  while (__int_as_float(old) < val) {
    int assumed = old;
    old = atomicCAS(ai, assumed, __float_as_int(val));
    if (old == assumed) break;
  }
}

// ---------------------------------------------------------------- cvt / init
__global__ void cvt_kernel(const float* __restrict__ in, half_t* __restrict__ out, int n) {
  int i = blockIdx.x * 256 + threadIdx.x;
  int stride = gridDim.x * 256;
  for (; i < n; i += stride) out[i] = (half_t)in[i];
}

__global__ void init_gmax_kernel(float* g) { *g = -3.0e38f; }

// Transpose + downconvert a weight: in[K][N] f32 -> out[N][K] f16.
// 32x32 tiles, 256 threads: coalesced loads, vectorized stores.
__global__ __launch_bounds__(256)
void transpose_w_kernel(const float* __restrict__ in, half_t* __restrict__ out,
                        int K, int N) {
  __shared__ half_t Tsh[32][36];
  const int tid = threadIdx.x;
  const int k0 = blockIdx.y * 32, n0 = blockIdx.x * 32;
  const int r = tid >> 3, c4 = (tid & 7) * 4;
  v4f f = *(const v4f*)(in + (size_t)(k0 + r) * N + n0 + c4);
#pragma unroll
  for (int i = 0; i < 4; i++) Tsh[c4 + i][r] = (half_t)f[i];
  __syncthreads();
  v4h o;
#pragma unroll
  for (int i = 0; i < 4; i++) o[i] = Tsh[r][c4 + i];
  *(v4h*)(out + (size_t)(n0 + r) * K + k0 + c4) = o;
}

// ---------------------------------------------------------------- generic WMMA GEMM
// C[M,N] = A[M,K] * B[K,N], with B supplied TRANSPOSED as Bt[N,K] (f16 weights).
// Block tile 256x128, BK=32; 8 waves, each wave owns 64x64 = 4x4 WMMA tiles.
enum { EPI_F32 = 0, EPI_F16 = 1, EPI_BIAS_ELU_F16 = 2, EPI_BIAS_F32 = 3 };

template <int EPI>
__global__ __launch_bounds__(256)
void gemm_f16(const half_t* __restrict__ A, const half_t* __restrict__ Bt,
              const float* __restrict__ bias, float* __restrict__ Cf,
              half_t* __restrict__ Ch, int Mdim, int Ndim, int Kdim) {
  constexpr int BM = 256, BN = 128, BK = 32, LD = BK + 8;
  __shared__ half_t Ash[BM][LD];  // [m][k]
  __shared__ half_t Bsh[BN][LD];  // [n][k]
  const int tid = threadIdx.x, wave = tid >> 5, lane = tid & 31;
  const int l16 = lane & 15, lhi = lane >> 4, k0 = lhi * 8;
  const int bm = blockIdx.y * BM, bn = blockIdx.x * BN;
  const int wm = (wave >> 1) * 64, wn = (wave & 1) * 64;

  const v8f vzero = {0.f, 0.f, 0.f, 0.f, 0.f, 0.f, 0.f, 0.f};
  v8f acc[4][4];
#pragma unroll
  for (int i = 0; i < 4; i++)
#pragma unroll
    for (int j = 0; j < 4; j++) acc[i][j] = vzero;

  for (int kb = 0; kb < Kdim; kb += BK) {
    {  // A tile 256x32: thread t stages its whole row (4 x 16B)
      const half_t* src = A + (size_t)(bm + tid) * Kdim + kb;
      half_t* dst = &Ash[tid][0];
      copy16_g2l(src + 0,  dst + 0);
      copy16_g2l(src + 8,  dst + 8);
      copy16_g2l(src + 16, dst + 16);
      copy16_g2l(src + 24, dst + 24);
    }
    {  // Bt tile 128x32: thread t stages a half-row (2 x 16B)
      int row = tid >> 1, seg = tid & 1;
      const half_t* src = Bt + (size_t)(bn + row) * Kdim + kb + seg * 16;
      half_t* dst = &Bsh[row][seg * 16];
      copy16_g2l(src + 0, dst + 0);
      copy16_g2l(src + 8, dst + 8);
    }
    async_wait_all();
    __syncthreads();

    v16h bf[4];
#pragma unroll
    for (int j = 0; j < 4; j++) {
      const half_t* p = &Bsh[wn + j * 16 + l16][0];
      bf[j] = mkfrag(*(const v8h*)(p + k0), *(const v8h*)(p + k0 + 16));
    }
#pragma unroll
    for (int i = 0; i < 4; i++) {
      const half_t* p = &Ash[wm + i * 16 + l16][0];
      v16h af = mkfrag(*(const v8h*)(p + k0), *(const v8h*)(p + k0 + 16));
#pragma unroll
      for (int j = 0; j < 4; j++) acc[i][j] = wmma32(af, bf[j], acc[i][j]);
    }
    __syncthreads();
  }

#pragma unroll
  for (int i = 0; i < 4; i++)
#pragma unroll
    for (int j = 0; j < 4; j++) {
      int col = bn + wn + j * 16 + l16;
      int rbase = bm + wm + i * 16 + lhi * 8;
      float bv = 0.f;
      if (EPI == EPI_BIAS_ELU_F16 || EPI == EPI_BIAS_F32) bv = bias[col];
#pragma unroll
      for (int r = 0; r < 8; r++) {
        float vv = acc[i][j][r] + bv;
        if (EPI == EPI_BIAS_ELU_F16) vv = vv > 0.f ? vv : (__expf(vv) - 1.f);
        size_t idx = (size_t)(rbase + r) * Ndim + col;
        if (EPI == EPI_F32 || EPI == EPI_BIAS_F32) Cf[idx] = vv;
        else                                        Ch[idx] = (half_t)vv;
      }
    }
}

// ---------------------------------------------------------------- FAVOR+ features
// data rows are (b,n,h) of length 64.  MODE: 0=qf (row max), 1=k global-max pass,
// 2=kf (uses gmax).  4 rows per block, 64 threads per row.
template <int MODE>
__global__ __launch_bounds__(256)
void feat_kernel(const half_t* __restrict__ data, const float* __restrict__ proj,
                 float* __restrict__ gmax, half_t* __restrict__ out) {
  __shared__ float projs[64][65];
  __shared__ float qs[4][64];
  __shared__ float dds[4][64];
  const int tid = threadIdx.x;
  const int g = tid >> 6, m = tid & 63;
  const size_t rowid = (size_t)blockIdx.x * 4 + g;

#pragma unroll
  for (int i = 0; i < 16; i++) {
    int idx = i * 256 + tid;
    projs[idx >> 6][idx & 63] = proj[idx];
  }
  qs[g][m] = (float)data[rowid * 64 + m];
  __syncthreads();

  const float dn = 0.3535533905932738f;  // 64^-0.25
  float dd = 0.f, s2 = 0.f;
#pragma unroll 8
  for (int d = 0; d < 64; d++) {
    float q = qs[g][d];
    dd += q * projs[m][d];
    s2 += q * q;
  }
  dd *= dn;
  float diag = s2 * dn * dn * 0.5f;

  if (MODE == 1) {
    float* red = &dds[0][0];
    red[tid] = dd;
    __syncthreads();
    for (int s = 128; s > 0; s >>= 1) {
      if (tid < s) red[tid] = fmaxf(red[tid], red[tid + s]);
      __syncthreads();
    }
    if (tid == 0) atomicMaxF(gmax, red[0]);
  } else {
    float mx;
    if (MODE == 0) {
      dds[g][m] = dd;
      __syncthreads();
      mx = -1e30f;
#pragma unroll 8
      for (int i = 0; i < 64; i++) mx = fmaxf(mx, dds[g][i]);
    } else {
      mx = *gmax;
    }
    float r = 0.125f * (__expf(dd - diag - mx) + 1e-6f);
    out[rowid * 64 + m] = (half_t)r;
  }
}

// ---------------------------------------------------------------- kv / ksum
// Per (b,h): kv[64m,64d] = sum_n kf[n,m]*v[n,d]; ksum[m] = sum_n kf[n,m].
__global__ __launch_bounds__(256)
void kv_ksum_kernel(const half_t* __restrict__ kf, const half_t* __restrict__ v,
                    float* __restrict__ kv, float* __restrict__ ksum) {
  constexpr int LD = 40;
  __shared__ half_t Kfs[64][LD];  // [m][n]
  __shared__ half_t Vs[64][LD];   // [d][n]
  const int bh = blockIdx.x, b = bh >> 4, h = bh & 15;
  const int tid = threadIdx.x, wave = tid >> 5, lane = tid & 31;
  const int l16 = lane & 15, lhi = lane >> 4, k0 = lhi * 8;
  const int tr = wave >> 1, tca = (wave & 1) * 2;

  const v8f vzero = {0.f, 0.f, 0.f, 0.f, 0.f, 0.f, 0.f, 0.f};
  v8f acc[2] = {vzero, vzero};
  float msum = 0.f;

  for (int n0 = 0; n0 < SEQ; n0 += 32) {
    int nn = tid >> 3, seg = tid & 7;
    size_t rbase = ((size_t)(b * SEQ + n0 + nn) * NHEAD + h) << 6;
    v8h tk = *(const v8h*)(kf + rbase + seg * 8);
    v8h tv = *(const v8h*)(v + rbase + seg * 8);
#pragma unroll
    for (int i = 0; i < 8; i++) {
      Kfs[seg * 8 + i][nn] = tk[i];
      Vs[seg * 8 + i][nn]  = tv[i];
    }
    __syncthreads();

    const half_t* pa = &Kfs[tr * 16 + l16][0];
    v16h af = mkfrag(*(const v8h*)(pa + k0), *(const v8h*)(pa + k0 + 16));
#pragma unroll
    for (int j = 0; j < 2; j++) {
      const half_t* pb = &Vs[(tca + j) * 16 + l16][0];
      v16h bf = mkfrag(*(const v8h*)(pb + k0), *(const v8h*)(pb + k0 + 16));
      acc[j] = wmma32(af, bf, acc[j]);
    }
    if (tid < 64) {
#pragma unroll
      for (int q = 0; q < 32; q++) msum += (float)Kfs[tid][q];
    }
    __syncthreads();
  }

#pragma unroll
  for (int j = 0; j < 2; j++) {
    int col = (tca + j) * 16 + l16;
    int rbase = tr * 16 + lhi * 8;
#pragma unroll
    for (int r = 0; r < 8; r++)
      kv[((size_t)bh * 64 + rbase + r) * 64 + col] = acc[j][r];
  }
  if (tid < 64) ksum[bh * 64 + tid] = msum;
}

// ---------------------------------------------------------------- num/den/attn
// Per (b,h, 128-row tile): num = qf(128x64m) * kv(64m x 64d), den = qf . ksum,
// attn = num/den -> f16 [BN, H*DH]
__global__ __launch_bounds__(256)
void attn_kernel(const half_t* __restrict__ qf, const float* __restrict__ kv,
                 const float* __restrict__ ksum, half_t* __restrict__ attn) {
  constexpr int LD = 72;
  __shared__ half_t Qs[128][LD];  // [n'][m]
  __shared__ half_t Ks[64][LD];   // [d][m]
  __shared__ float dsh[128];
  __shared__ float kss[64];
  const int bh = blockIdx.y, b = bh >> 4, h = bh & 15;
  const int n0 = blockIdx.x * 128;
  const int tid = threadIdx.x, wave = tid >> 5, lane = tid & 31;
  const int l16 = lane & 15, lhi = lane >> 4, k0 = lhi * 8;

  {  // qf tile
    int row = tid >> 1, seg = tid & 1;
    const half_t* p = qf + (((size_t)(b * SEQ + n0 + row) * NHEAD + h) << 6) + seg * 32;
#pragma unroll
    for (int i = 0; i < 4; i++) *(v8h*)&Qs[row][seg * 32 + i * 8] = ((const v8h*)p)[i];
  }
#pragma unroll
  for (int i = 0; i < 16; i++) {  // kv -> Ks[d][m] f16
    int idx = tid * 16 + i;
    int m = idx >> 6, d = idx & 63;
    Ks[d][m] = (half_t)kv[((size_t)bh * 64 + m) * 64 + d];
  }
  if (tid < 64) kss[tid] = ksum[bh * 64 + tid];
  __syncthreads();

  if (tid < 128) {
    float s = 0.f;
#pragma unroll 8
    for (int m = 0; m < 64; m++) s += (float)Qs[tid][m] * kss[m];
    dsh[tid] = s;
  }
  __syncthreads();

  const v8f vzero = {0.f, 0.f, 0.f, 0.f, 0.f, 0.f, 0.f, 0.f};
  v8f acc[4] = {vzero, vzero, vzero, vzero};
#pragma unroll
  for (int m0 = 0; m0 < 64; m0 += 32) {
    const half_t* pa = &Qs[wave * 16 + l16][m0];
    v16h af = mkfrag(*(const v8h*)(pa + k0), *(const v8h*)(pa + k0 + 16));
#pragma unroll
    for (int j = 0; j < 4; j++) {
      const half_t* pb = &Ks[j * 16 + l16][m0];
      v16h bf = mkfrag(*(const v8h*)(pb + k0), *(const v8h*)(pb + k0 + 16));
      acc[j] = wmma32(af, bf, acc[j]);
    }
  }

#pragma unroll
  for (int j = 0; j < 4; j++) {
    int d = j * 16 + l16;
    int rbase = wave * 16 + lhi * 8;
#pragma unroll
    for (int r = 0; r < 8; r++) {
      int rl = rbase + r;
      float vv = acc[j][r] / dsh[rl];
      attn[(((size_t)(b * SEQ + n0 + rl) * NHEAD + h) << 6) + d] = (half_t)vv;
    }
  }
}

// ---------------------------------------------------------------- residual + LN
template <bool WRITE_H>
__global__ __launch_bounds__(256)
void add_ln_kernel(const float* __restrict__ in1, const float* __restrict__ in2,
                   const float* __restrict__ gamma, const float* __restrict__ beta,
                   float* __restrict__ out, half_t* __restrict__ outh) {
  __shared__ float red[256];
  __shared__ float stats[2];
  const int row = blockIdx.x, tid = threadIdx.x;
  const size_t base = (size_t)row * DMODEL;
  float s[4];
  float psum = 0.f;
#pragma unroll
  for (int i = 0; i < 4; i++) {
    int c = tid + i * 256;
    s[i] = in1[base + c] + in2[base + c];
    psum += s[i];
  }
  red[tid] = psum;
  __syncthreads();
  for (int st = 128; st > 0; st >>= 1) {
    if (tid < st) red[tid] += red[tid + st];
    __syncthreads();
  }
  if (tid == 0) stats[0] = red[0] * (1.f / DMODEL);
  __syncthreads();
  float mu = stats[0];
  float pv = 0.f;
#pragma unroll
  for (int i = 0; i < 4; i++) { float d = s[i] - mu; pv += d * d; }
  red[tid] = pv;
  __syncthreads();
  for (int st = 128; st > 0; st >>= 1) {
    if (tid < st) red[tid] += red[tid + st];
    __syncthreads();
  }
  if (tid == 0) stats[1] = red[0] * (1.f / DMODEL);
  __syncthreads();
  float rs = rsqrtf(stats[1] + 1e-6f);
#pragma unroll
  for (int i = 0; i < 4; i++) {
    int c = tid + i * 256;
    float y = (s[i] - mu) * rs * gamma[c] + beta[c];
    out[base + c] = y;
    if (WRITE_H) outh[base + c] = (half_t)y;
  }
}

// ---------------------------------------------------------------- launch
extern "C" void kernel_launch(void* const* d_in, const int* in_sizes, int n_in,
                              void* d_out, int out_size, void* d_ws, size_t ws_size,
                              hipStream_t stream) {
  (void)in_sizes; (void)n_in; (void)out_size; (void)ws_size;
  const float* x    = (const float*)d_in[0];
  const float* Wq   = (const float*)d_in[1];
  const float* Wk   = (const float*)d_in[2];
  const float* Wv   = (const float*)d_in[3];
  const float* Wo   = (const float*)d_in[4];
  const float* proj = (const float*)d_in[5];
  const float* W1   = (const float*)d_in[6];
  const float* b1   = (const float*)d_in[7];
  const float* W2   = (const float*)d_in[8];
  const float* b2   = (const float*)d_in[9];
  const float* ln1g = (const float*)d_in[10];
  const float* ln1b = (const float*)d_in[11];
  const float* ln2g = (const float*)d_in[12];
  const float* ln2b = (const float*)d_in[13];
  float* out = (float*)d_out;

  char* ws = (char*)d_ws;
  size_t off = 0;
  auto alloc = [&](size_t bytes) -> char* {
    char* p = ws + off;
    off += (bytes + 255) & ~(size_t)255;
    return p;
  };
  const size_t NT = (size_t)BATCH * SEQ;  // 16384 tokens

  half_t* xh   = (half_t*)alloc(NT * DMODEL * 2);
  half_t* wqt  = (half_t*)alloc((size_t)DMODEL * DMODEL * 2);  // transposed f16
  half_t* wkt  = (half_t*)alloc((size_t)DMODEL * DMODEL * 2);
  half_t* wvt  = (half_t*)alloc((size_t)DMODEL * DMODEL * 2);
  half_t* wot  = (half_t*)alloc((size_t)DMODEL * DMODEL * 2);
  half_t* w1t  = (half_t*)alloc((size_t)DMODEL * DFF * 2);
  half_t* w2t  = (half_t*)alloc((size_t)DFF * DMODEL * 2);
  half_t* qh   = (half_t*)alloc(NT * DMODEL * 2);  // q,k,v,kf: contiguous 4x32MB,
  half_t* kh   = (half_t*)alloc(NT * DMODEL * 2);  // recycled as the [NT,DFF] f16
  half_t* vh   = (half_t*)alloc(NT * DMODEL * 2);  // ELU activation buffer
  half_t* kfh  = (half_t*)alloc(NT * DMODEL * 2);
  half_t* heh  = qh;  (void)kfh;
  half_t* qfh  = (half_t*)alloc(NT * DMODEL * 2);
  float* kvb   = (float*)alloc((size_t)BATCH * NHEAD * MFEAT * DHEAD * 4);
  float* ksumb = (float*)alloc((size_t)BATCH * NHEAD * MFEAT * 4);
  float* gmaxb = (float*)alloc(256);
  half_t* attnh = (half_t*)alloc(NT * DMODEL * 2);
  float* aout  = (float*)alloc(NT * DMODEL * 4);
  float* ffn   = aout;            // reuse: attn_out dead after LN1
  float* out1  = (float*)alloc(NT * DMODEL * 4);
  half_t* out1h = xh;             // reuse: xh dead after QKV GEMMs

  dim3 blk(256);

  // activations f32 -> f16; weights f32 -> f16 transposed [N,K]
  cvt_kernel<<<4096, blk, 0, stream>>>(x, xh, (int)(NT * DMODEL));
  {
    dim3 gt(DMODEL / 32, DMODEL / 32);
    transpose_w_kernel<<<gt, blk, 0, stream>>>(Wq, wqt, DMODEL, DMODEL);
    transpose_w_kernel<<<gt, blk, 0, stream>>>(Wk, wkt, DMODEL, DMODEL);
    transpose_w_kernel<<<gt, blk, 0, stream>>>(Wv, wvt, DMODEL, DMODEL);
    transpose_w_kernel<<<gt, blk, 0, stream>>>(Wo, wot, DMODEL, DMODEL);
    dim3 g1t(DFF / 32, DMODEL / 32);   // W1[K=DMODEL][N=DFF]
    transpose_w_kernel<<<g1t, blk, 0, stream>>>(W1, w1t, DMODEL, DFF);
    dim3 g2t(DMODEL / 32, DFF / 32);   // W2[K=DFF][N=DMODEL]
    transpose_w_kernel<<<g2t, blk, 0, stream>>>(W2, w2t, DFF, DMODEL);
  }
  init_gmax_kernel<<<1, 1, 0, stream>>>(gmaxb);

  // QKV projections (WMMA, 256x128 tiles)
  dim3 g1(DMODEL / 128, (unsigned)(NT / 256));
  gemm_f16<EPI_F16><<<g1, blk, 0, stream>>>(xh, wqt, nullptr, nullptr, qh,
                                            (int)NT, DMODEL, DMODEL);
  gemm_f16<EPI_F16><<<g1, blk, 0, stream>>>(xh, wkt, nullptr, nullptr, kh,
                                            (int)NT, DMODEL, DMODEL);
  gemm_f16<EPI_F16><<<g1, blk, 0, stream>>>(xh, wvt, nullptr, nullptr, vh,
                                            (int)NT, DMODEL, DMODEL);

  // FAVOR+ feature maps
  int frows = (int)(NT * NHEAD / 4);  // 65536 blocks
  feat_kernel<0><<<frows, blk, 0, stream>>>(qh, proj, nullptr, qfh);
  feat_kernel<1><<<frows, blk, 0, stream>>>(kh, proj, gmaxb, nullptr);
  feat_kernel<2><<<frows, blk, 0, stream>>>(kh, proj, gmaxb, kfh);

  // linear attention (WMMA)
  kv_ksum_kernel<<<BATCH * NHEAD, blk, 0, stream>>>(kfh, vh, kvb, ksumb);
  dim3 ga(SEQ / 128, BATCH * NHEAD);
  attn_kernel<<<ga, blk, 0, stream>>>(qfh, kvb, ksumb, attnh);

  // output projection + LN1
  gemm_f16<EPI_F32><<<g1, blk, 0, stream>>>(attnh, wot, nullptr, aout, nullptr,
                                            (int)NT, DMODEL, DMODEL);
  add_ln_kernel<true><<<(unsigned)NT, blk, 0, stream>>>(x, aout, ln1g, ln1b,
                                                        out1, out1h);

  // FFN (WMMA) + LN2
  dim3 g2(DFF / 128, (unsigned)(NT / 256));
  gemm_f16<EPI_BIAS_ELU_F16><<<g2, blk, 0, stream>>>(out1h, w1t, b1, nullptr, heh,
                                                     (int)NT, DFF, DMODEL);
  gemm_f16<EPI_BIAS_F32><<<g1, blk, 0, stream>>>(heh, w2t, b2, ffn, nullptr,
                                                 (int)NT, DMODEL, DFF);
  add_ln_kernel<false><<<(unsigned)NT, blk, 0, stream>>>(out1, ffn, ln2g, ln2b,
                                                         out, nullptr);
}
